// DynamicSparseAttention_69535520522270
// MI455X (gfx1250) — matrix operand order repaired and checked by reading, not compiled, not run on hardware
//
#include <hip/hip_runtime.h>
#include <hip/hip_bf16.h>

// ---------------------------------------------------------------------------
// DynamicSparseAttention on MI455X (gfx1250): bf16 WMMA pipeline.
//   feat = mean+max -> silu MLP -> sigmoid gate            (scalar kernels)
//   Q=(n,c)  K=(n,c)  V=(c,n)  all bf16 in workspace       (WMMA GEMMs, 32x32/wave)
//   flash-attention: K/V tiles staged to LDS per block via
//     GLOBAL_LOAD_ASYNC_TO_LDS_B128 (ASYNCcnt), WMMA S and PV (16 q-rows/wave)
//   out = x + (Wproj @ attn^T) * gate                      (WMMA GEMM, fused)
// Fragment layouts used here (A/B in this code's contiguous k order, not the ISA table order):
//   A 16x32 bf16 : lane l -> row (l&15), K = (l>>4)*16 + {0..15}  (contiguous)
//   B 32x16 bf16 : lane l -> col (l&15), K = (l>>4)*16 + {0..15}  (contiguous)
//   C/D 16x16 f32: VGPR r -> row r + 8*(l>>4), col (l&15)
// ---------------------------------------------------------------------------

typedef __attribute__((ext_vector_type(16))) __bf16 bf16x16;
typedef __attribute__((ext_vector_type(8)))  __bf16 bf16x8;
typedef __attribute__((ext_vector_type(8)))  float  f32x8;
typedef int v4i __attribute__((vector_size(16)));   // matches builtin prototype

#define WMMA_BF16(A, Bm, Cm) \
  __builtin_amdgcn_wmma_f32_16x16x32_bf16(false, (A), false, (Bm), (short)0, (Cm), false, false)

// 16-byte async DMA global -> LDS (tracked by ASYNCcnt); safe fallback.
__device__ __forceinline__ void async_copy16(const __bf16* g, __bf16* l) {
#if __has_builtin(__builtin_amdgcn_global_load_async_to_lds_b128)
  __builtin_amdgcn_global_load_async_to_lds_b128(
      (__attribute__((address_space(1))) v4i*)(v4i*)g,
      (__attribute__((address_space(3))) v4i*)(v4i*)l, 0, 0);
#else
  *(bf16x8*)l = *(const bf16x8*)g;
#endif
}

__device__ __forceinline__ void wait_async0() {
#if __has_builtin(__builtin_amdgcn_s_wait_asynccnt)
  __builtin_amdgcn_s_wait_asynccnt(0);
#else
  asm volatile("s_wait_asynccnt 0" ::: "memory");
#endif
}

// A fragment (M x K, row-major, K contiguous): two 16B loads per lane.
__device__ __forceinline__ bf16x16 frag_a_global(const __bf16* A, int lda, int m0, int k0, int lane) {
  const int m   = m0 + (lane & 15);
  const int sel = lane >> 4;
  const __bf16* p = A + (size_t)m * lda + k0 + sel * 16;
  bf16x8 lo = *(const bf16x8*)(p);
  bf16x8 hi = *(const bf16x8*)(p + 8);
  bf16x16 r;
#pragma unroll
  for (int i = 0; i < 8; ++i) { r[i] = lo[i]; r[8 + i] = hi[i]; }
  return r;
}

// B fragment from (N x K, row-major, K contiguous): one 32B (2x16B) load per lane.
__device__ __forceinline__ bf16x16 frag_b_global(const __bf16* B, int ldb, int n0, int k0, int lane) {
  const int n   = n0 + (lane & 15);
  const int sel = lane >> 4;
  const __bf16* p = B + (size_t)n * ldb + k0 + sel * 16;
  bf16x8 lo = *(const bf16x8*)(p);
  bf16x8 hi = *(const bf16x8*)(p + 8);
  bf16x16 r;
#pragma unroll
  for (int i = 0; i < 8; ++i) { r[i] = lo[i]; r[8 + i] = hi[i]; }
  return r;
}

// B fragment from a K-contiguous 32B chunk (used for LDS tiles).
__device__ __forceinline__ bf16x16 frag_pair(const __bf16* p) {
  bf16x8 lo = *(const bf16x8*)(p);
  bf16x8 hi = *(const bf16x8*)(p + 8);
  bf16x16 r;
#pragma unroll
  for (int i = 0; i < 8; ++i) { r[i] = lo[i]; r[8 + i] = hi[i]; }
  return r;
}

// ---------------------------------------------------------------------------
// Generic WMMA GEMM: O(m,n) = sum_k A[m,k]*B[n,k].
// 8 waves/block, each wave owns a 32x32 tile (2x2 WMMA, 2x fragment reuse).
// Block tile: 64(M) x 128(N).
// mode 0: bf16 store. mode 1: f32 out = x + val*gate (fused residual epilogue).
// ---------------------------------------------------------------------------
__device__ __forceinline__ void gemm_store(void* Out, int mode, int bz, int mt, int nt,
                                           int ldo, size_t oStride,
                                           const float* __restrict__ xres,
                                           const float* __restrict__ cplx,
                                           const f32x8& acc, int lane) {
  const int col   = nt + (lane & 15);
  const int rbase = mt + (lane >> 4) * 8;
  if (mode == 0) {
    __bf16* O = (__bf16*)Out + (size_t)bz * oStride;
#pragma unroll
    for (int r = 0; r < 8; ++r)
      O[(size_t)(rbase + r) * ldo + col] = (__bf16)acc[r];
  } else {
    float* O = (float*)Out;
    const float g = cplx[bz];
    const size_t base = (size_t)bz * oStride;
#pragma unroll
    for (int r = 0; r < 8; ++r) {
      size_t idx = base + (size_t)(rbase + r) * ldo + col;
      O[idx] = xres[idx] + acc[r] * g;
    }
  }
}

__global__ __launch_bounds__(256)
void wmma_gemm_kernel(const __bf16* __restrict__ A, const __bf16* __restrict__ B,
                      void* __restrict__ Out,
                      int K, int lda, int ldb, int ldo,
                      size_t aStride, size_t bStride, size_t oStride,
                      int mode, const float* __restrict__ xres, const float* __restrict__ cplx) {
  const int lane = threadIdx.x & 31;
  const int wave = threadIdx.x >> 5;
  const int wm = wave >> 2;   // 0..1
  const int wn = wave & 3;    // 0..3
  const int bz = blockIdx.z;
  const int m0 = blockIdx.x * 64  + wm * 32;
  const int n0 = blockIdx.y * 128 + wn * 32;
  A += (size_t)bz * aStride;
  B += (size_t)bz * bStride;

  f32x8 acc00 = {}, acc01 = {}, acc10 = {}, acc11 = {};
  for (int k0 = 0; k0 < K; k0 += 32) {
    bf16x16 a0 = frag_a_global(A, lda, m0,      k0, lane);
    bf16x16 a1 = frag_a_global(A, lda, m0 + 16, k0, lane);
    bf16x16 b0 = frag_b_global(B, ldb, n0,      k0, lane);
    bf16x16 b1 = frag_b_global(B, ldb, n0 + 16, k0, lane);
    acc00 = WMMA_BF16(a0, b0, acc00);
    acc01 = WMMA_BF16(a0, b1, acc01);
    acc10 = WMMA_BF16(a1, b0, acc10);
    acc11 = WMMA_BF16(a1, b1, acc11);
  }
  gemm_store(Out, mode, bz, m0,      n0,      ldo, oStride, xres, cplx, acc00, lane);
  gemm_store(Out, mode, bz, m0,      n0 + 16, ldo, oStride, xres, cplx, acc01, lane);
  gemm_store(Out, mode, bz, m0 + 16, n0,      ldo, oStride, xres, cplx, acc10, lane);
  gemm_store(Out, mode, bz, m0 + 16, n0 + 16, ldo, oStride, xres, cplx, acc11, lane);
}

// ---------------------------------------------------------------------------
// Flash attention: block = 4 waves = 64 query rows; keys streamed in chunks
// of 32 through LDS tiles loaded with async DMA and shared by all waves.
// Q,K stored (n,c) bf16; V stored (c,n) bf16; O stored (n,c) bf16.
// LDS: K tile 16KB + V tile 16KB + P tiles 4KB = 36KB / block.
// ---------------------------------------------------------------------------
__global__ __launch_bounds__(128)
void flash_attn_kernel(const __bf16* __restrict__ Q, const __bf16* __restrict__ Kb,
                       const __bf16* __restrict__ V, __bf16* __restrict__ O,
                       int Ntok, int C, float scale) {
  __shared__ __bf16 kTile[32 * 256];   // (key, c) row-major, K-contiguous
  __shared__ __bf16 vTile[256 * 32];   // (c, key) row-major, key-contiguous
  __shared__ __bf16 pTile[4][16 * 32]; // per-wave P reshape buffer

  const int tid  = threadIdx.x;
  const int lane = tid & 31;
  const int wave = tid >> 5;
  const int bz   = blockIdx.z;
  const int q0   = (blockIdx.x * 4 + wave) * 16;
  const int sel  = lane >> 4;
  const int ln   = lane & 15;

  const size_t bs = (size_t)Ntok * C;
  Q += bz * bs;  Kb += bz * bs;  V += bz * bs;  O += bz * bs;

  f32x8 oacc[16];
  f32x8 zero = {};
#pragma unroll
  for (int f = 0; f < 16; ++f) oacc[f] = zero;
  float mrun[8], lrun[8];
#pragma unroll
  for (int r = 0; r < 8; ++r) { mrun[r] = -3.0e38f; lrun[r] = 0.0f; }

  for (int j0 = 0; j0 < Ntok; j0 += 32) {
    // ---- cooperative async DMA of K/V tiles into LDS (16 issues/thread) ----
    {
      const __bf16* gk = Kb + (size_t)j0 * C;  // 32 rows x 256 c = 16KB contiguous
#pragma unroll
      for (int i = 0; i < 8; ++i) {
        int idx = tid + i * 128;               // 1024 x 16B chunks
        async_copy16(gk + idx * 8, &kTile[idx * 8]);
      }
#pragma unroll
      for (int i = 0; i < 8; ++i) {
        int idx = tid + i * 128;               // 256 rows x 4 chunks
        int c = idx >> 2, qc = idx & 3;
        async_copy16(V + (size_t)c * Ntok + j0 + qc * 8, &vTile[c * 32 + qc * 8]);
      }
    }
    wait_async0();
    __syncthreads();

    // ---- S tile: 16 queries x 32 keys, K-dim = C (B-frags from LDS) ----
    f32x8 s0 = zero, s1 = zero;
    for (int k0 = 0; k0 < C; k0 += 32) {
      bf16x16 aq = frag_a_global(Q, C, q0, k0, lane);
      bf16x16 b0 = frag_pair(&kTile[ln * 256 + k0 + sel * 16]);
      bf16x16 b1 = frag_pair(&kTile[(16 + ln) * 256 + k0 + sel * 16]);
      s0 = WMMA_BF16(aq, b0, s0);
      s1 = WMMA_BF16(aq, b1, s1);
    }

    // ---- online softmax (row r at r + 8*sel; cols across 16-lane half) ----
    float alpha[8];
#pragma unroll
    for (int r = 0; r < 8; ++r) {
      float v0 = s0[r] * scale, v1 = s1[r] * scale;
      float rm = fmaxf(v0, v1);
#pragma unroll
      for (int msk = 1; msk < 16; msk <<= 1)
        rm = fmaxf(rm, __shfl_xor(rm, msk, 16));
      float mnew = fmaxf(mrun[r], rm);
      alpha[r] = __expf(mrun[r] - mnew);
      float p0 = __expf(v0 - mnew);
      float p1 = __expf(v1 - mnew);
      s0[r] = p0; s1[r] = p1;
      float sum = p0 + p1;
#pragma unroll
      for (int msk = 1; msk < 16; msk <<= 1)
        sum += __shfl_xor(sum, msk, 16);
      lrun[r] = lrun[r] * alpha[r] + sum;
      mrun[r] = mnew;
    }
#pragma unroll
    for (int f = 0; f < 16; ++f)
#pragma unroll
      for (int r = 0; r < 8; ++r) oacc[f][r] *= alpha[r];

    // ---- reshape P (C/D layout) -> A fragment via per-wave LDS tile ----
#pragma unroll
    for (int r = 0; r < 8; ++r) {
      const int row = r + 8 * sel;
      pTile[wave][row * 32 + ln]      = (__bf16)s0[r];
      pTile[wave][row * 32 + 16 + ln] = (__bf16)s1[r];
    }
    asm volatile("s_wait_dscnt 0" ::: "memory");
    bf16x16 ap;
    {
      const __bf16* p = &pTile[wave][ln * 32 + sel * 16];
      bf16x8 lo = *(const bf16x8*)(p);
      bf16x8 hi = *(const bf16x8*)(p + 8);
#pragma unroll
      for (int i = 0; i < 8; ++i) { ap[i] = lo[i]; ap[8 + i] = hi[i]; }
    }

    // ---- O += P @ V  (V B-frags from LDS, keys contiguous per feature) ----
#pragma unroll 4
    for (int f = 0; f < 16; ++f) {
      bf16x16 bv = frag_pair(&vTile[(f * 16 + ln) * 32 + sel * 16]);
      oacc[f] = WMMA_BF16(ap, bv, oacc[f]);
    }
    __syncthreads();   // all waves done with kTile/vTile before next DMA
  }

  // ---- normalize and store O (n,c) bf16 ----
  float inv[8];
#pragma unroll
  for (int r = 0; r < 8; ++r) inv[r] = 1.0f / lrun[r];
#pragma unroll
  for (int f = 0; f < 16; ++f)
#pragma unroll
    for (int r = 0; r < 8; ++r) {
      const int row = q0 + r + 8 * sel;
      O[(size_t)row * C + f * 16 + ln] = (__bf16)(oacc[f][r] * inv[r]);
    }
}

// ---------------------------------------------------------------------------
// Small helper kernels
// ---------------------------------------------------------------------------
__global__ __launch_bounds__(256)
void feat_kernel(const float* __restrict__ x, float* __restrict__ feat, int HW, int C) {
  const int c = blockIdx.x, b = blockIdx.y;
  const float* p = x + ((size_t)b * C + c) * HW;
  float s = 0.f, mx = -3.0e38f;
  for (int i = threadIdx.x; i < HW; i += 256) {
    float v = p[i];
    s += v; mx = fmaxf(mx, v);
  }
  __shared__ float ss[256], sm[256];
  ss[threadIdx.x] = s; sm[threadIdx.x] = mx;
  __syncthreads();
  for (int st = 128; st > 0; st >>= 1) {
    if (threadIdx.x < st) {
      ss[threadIdx.x] += ss[threadIdx.x + st];
      sm[threadIdx.x] = fmaxf(sm[threadIdx.x], sm[threadIdx.x + st]);
    }
    __syncthreads();
  }
  if (threadIdx.x == 0) feat[(size_t)b * C + c] = ss[0] / (float)HW + sm[0];
}

__global__ __launch_bounds__(128)
void gate_kernel(const float* __restrict__ feat, const float* __restrict__ w1,
                 const float* __restrict__ w2, float* __restrict__ cplx, int C, int HID) {
  __shared__ float hid[4][32];
  const int b = threadIdx.x >> 5, h = threadIdx.x & 31;
  float s = 0.f;
  for (int c = 0; c < C; ++c) s += w1[h * C + c] * feat[b * C + c];
  hid[b][h] = s / (1.0f + __expf(-s));   // silu
  __syncthreads();
  if (threadIdx.x < 4) {
    float t = 0.f;
    for (int hh = 0; hh < HID; ++hh) t += w2[hh] * hid[threadIdx.x][hh];
    cplx[threadIdx.x] = 1.0f / (1.0f + __expf(-t));  // sigmoid
  }
}

__global__ __launch_bounds__(256)
void cvt_w_kernel(const float* __restrict__ a, const float* __restrict__ b,
                  const float* __restrict__ c, const float* __restrict__ d,
                  __bf16* __restrict__ oa, __bf16* __restrict__ ob,
                  __bf16* __restrict__ oc, __bf16* __restrict__ od, int n) {
  int i = blockIdx.x * blockDim.x + threadIdx.x;
  if (i < n) { oa[i] = (__bf16)a[i]; ob[i] = (__bf16)b[i]; oc[i] = (__bf16)c[i]; od[i] = (__bf16)d[i]; }
}

// x (b,c,n) f32  ->  Xt (b,n,c) bf16  (32x32 LDS tile transpose)
__global__ __launch_bounds__(256)
void transpose_cvt_kernel(const float* __restrict__ x, __bf16* __restrict__ Xt, int C, int Ntok) {
  __shared__ float tile[32][33];
  const int b = blockIdx.z;
  const int n0 = blockIdx.x * 32, c0 = blockIdx.y * 32;
  const int tx = threadIdx.x, ty = threadIdx.y;  // 32 x 8
#pragma unroll
  for (int i = 0; i < 4; ++i) {
    int c = c0 + ty + i * 8;
    tile[ty + i * 8][tx] = x[((size_t)b * C + c) * Ntok + n0 + tx];
  }
  __syncthreads();
#pragma unroll
  for (int i = 0; i < 4; ++i) {
    int n = n0 + ty + i * 8;
    Xt[((size_t)b * Ntok + n) * C + c0 + tx] = (__bf16)tile[tx][ty + i * 8];
  }
}

// ---------------------------------------------------------------------------
extern "C" void kernel_launch(void* const* d_in, const int* in_sizes, int n_in,
                              void* d_out, int out_size, void* d_ws, size_t ws_size,
                              hipStream_t stream) {
  (void)in_sizes; (void)n_in; (void)out_size; (void)ws_size;
  const int B = 4, C = 256, H = 64, W = 64, HID = 32;
  const int N = H * W;  // 4096 tokens

  const float* x     = (const float*)d_in[0];
  const float* w_ce1 = (const float*)d_in[1];
  const float* w_ce2 = (const float*)d_in[2];
  const float* wq    = (const float*)d_in[3];
  const float* wk    = (const float*)d_in[4];
  const float* wv    = (const float*)d_in[5];
  const float* wproj = (const float*)d_in[6];
  float* out = (float*)d_out;

  // workspace carve-up (~42.6 MB total)
  char* ws = (char*)d_ws;
  size_t off = 0;
  auto carve = [&](size_t bytes) -> void* {
    void* p = ws + off;
    off += (bytes + 255) & ~(size_t)255;
    return p;
  };
  float*  feat = (float*)carve((size_t)B * C * sizeof(float));
  float*  cplx = (float*)carve((size_t)B * sizeof(float));
  __bf16* Wqb  = (__bf16*)carve((size_t)C * C * 2);
  __bf16* Wkb  = (__bf16*)carve((size_t)C * C * 2);
  __bf16* Wvb  = (__bf16*)carve((size_t)C * C * 2);
  __bf16* Wpb  = (__bf16*)carve((size_t)C * C * 2);
  __bf16* Xt   = (__bf16*)carve((size_t)B * N * C * 2);  // (b,n,c)
  __bf16* Qb   = (__bf16*)carve((size_t)B * N * C * 2);  // (b,n,c)
  __bf16* Kbf  = (__bf16*)carve((size_t)B * N * C * 2);  // (b,n,c)
  __bf16* Vb   = (__bf16*)carve((size_t)B * N * C * 2);  // (b,c,n)
  __bf16* Ob   = (__bf16*)carve((size_t)B * N * C * 2);  // (b,n,c)

  const size_t NC = (size_t)N * C;

  feat_kernel<<<dim3(C, B), 256, 0, stream>>>(x, feat, H * W, C);
  gate_kernel<<<1, 128, 0, stream>>>(feat, w_ce1, w_ce2, cplx, C, HID);
  cvt_w_kernel<<<(C * C + 255) / 256, 256, 0, stream>>>(wq, wk, wv, wproj, Wqb, Wkb, Wvb, Wpb, C * C);
  transpose_cvt_kernel<<<dim3(N / 32, C / 32, B), dim3(32, 8), 0, stream>>>(x, Xt, C, N);

  // Q(n,o) = sum_c Xt[n,c] * Wq[o,c]
  wmma_gemm_kernel<<<dim3(N / 64, C / 128, B), 256, 0, stream>>>(
      Xt, Wqb, Qb, C, C, C, C, NC, 0, NC, 0, nullptr, nullptr);
  // K(n,o)
  wmma_gemm_kernel<<<dim3(N / 64, C / 128, B), 256, 0, stream>>>(
      Xt, Wkb, Kbf, C, C, C, C, NC, 0, NC, 0, nullptr, nullptr);
  // V(o,n) = sum_c Wv[o,c] * Xt[n,c]
  wmma_gemm_kernel<<<dim3(C / 64, N / 128, B), 256, 0, stream>>>(
      Wvb, Xt, Vb, C, C, C, N, 0, NC, NC, 0, nullptr, nullptr);

  // attention: softmax(Q K^T / 16) V  -> Ob (n,c)
  flash_attn_kernel<<<dim3(N / 64, 1, B), 128, 0, stream>>>(
      Qb, Kbf, Vb, Ob, N, C, 1.0f / 16.0f);

  // out = x + (Wproj @ attn^T) * gate   (f32, fused epilogue)
  wmma_gemm_kernel<<<dim3(C / 64, N / 128, B), 256, 0, stream>>>(
      Wpb, Ob, out, C, C, C, N, 0, NC, NC, 1, x, cplx);
}
